// MultiheadAttention_30958124270124
// MI455X (gfx1250) — compile-verified, matrix operand order checked by reading, not run on hardware
//
#include <hip/hip_runtime.h>

typedef __attribute__((ext_vector_type(16))) _Float16 v16h;
typedef __attribute__((ext_vector_type(8)))  _Float16 f16x8;
typedef __attribute__((ext_vector_type(8)))  float    v8f;
typedef __attribute__((ext_vector_type(4)))  int      v4i;

#define MODEL_DIM 1024
#define NUM_HEAD  16
#define HEAD_DIM  64
#define BATCH     2
#define SEQ       2048
#define ROWS      (BATCH * SEQ)   // 4096

union AFrag { v16h v; f16x8 h[2]; };
union CFrag { v8f  v; float f[8]; };

// ---------------------------------------------------------------------------
// CDNA5 async global->LDS copy (ASYNCcnt path), with synchronous fallback.
// Probe-confirmed signature: (v4i addrspace(1)*, v4i addrspace(3)*, Ii, Ii)
// ---------------------------------------------------------------------------
#if defined(__HIP_DEVICE_COMPILE__) &&                                        \
    __has_builtin(__builtin_amdgcn_global_load_async_to_lds_b128) &&          \
    __has_builtin(__builtin_amdgcn_s_wait_asynccnt)
#define HAVE_ASYNC 1
#else
#define HAVE_ASYNC 0
#endif

__device__ __forceinline__ void cp16(_Float16* lds, const _Float16* g) {
#if HAVE_ASYNC
  __builtin_amdgcn_global_load_async_to_lds_b128(
      (__attribute__((address_space(1))) v4i*)(void*)g,
      (__attribute__((address_space(3))) v4i*)(void*)lds, 0, 0);
#else
  *(f16x8*)lds = *(const f16x8*)g;
#endif
}
__device__ __forceinline__ void cp_wait() {
#if HAVE_ASYNC
  __builtin_amdgcn_s_wait_asynccnt(0);
#endif
}

// ---------------------------------------------------------------------------
// One-time prep: f32 -> f16 elementwise (for q/k/v activations)
// ---------------------------------------------------------------------------
__global__ __launch_bounds__(256) void cvt_kernel(const float* __restrict__ X,
                                                  _Float16* __restrict__ Y) {
  size_t base = ((size_t)blockIdx.x * 256 + threadIdx.x) * 8;
  const float4 a = *(const float4*)(X + base);
  const float4 b = *(const float4*)(X + base + 4);
  f16x8 h;
  h[0] = (_Float16)a.x; h[1] = (_Float16)a.y; h[2] = (_Float16)a.z; h[3] = (_Float16)a.w;
  h[4] = (_Float16)b.x; h[5] = (_Float16)b.y; h[6] = (_Float16)b.z; h[7] = (_Float16)b.w;
  *(f16x8*)(Y + base) = h;
}

// ---------------------------------------------------------------------------
// One-time prep: W[k][n] f32  ->  WT[n][k] f16  (64x64 tiles through LDS)
// ---------------------------------------------------------------------------
__global__ __launch_bounds__(256) void wt_kernel(const float* __restrict__ W,
                                                 _Float16* __restrict__ WT) {
  __shared__ _Float16 tile[64][72];
  const int t  = threadIdx.x;
  const int n0 = blockIdx.x * 64, k0 = blockIdx.y * 64;
#pragma unroll
  for (int i = 0; i < 4; ++i) {
    int idx = t + i * 256;                       // 0..1023 float4 chunks
    int kk = idx >> 4, n4 = idx & 15;
    const float4 f = *(const float4*)(W + (size_t)(k0 + kk) * MODEL_DIM + n0 + n4 * 4);
    tile[n4 * 4 + 0][kk] = (_Float16)f.x;
    tile[n4 * 4 + 1][kk] = (_Float16)f.y;
    tile[n4 * 4 + 2][kk] = (_Float16)f.z;
    tile[n4 * 4 + 3][kk] = (_Float16)f.w;
  }
  __syncthreads();
#pragma unroll
  for (int i = 0; i < 2; ++i) {
    int idx = t + i * 256;                       // 0..511 f16x8 chunks
    int n = idx >> 3, c8 = idx & 7;
    *(f16x8*)(WT + (size_t)(n0 + n) * MODEL_DIM + k0 + c8 * 8) =
        *(const f16x8*)&tile[n][c8 * 8];
  }
}

// ---------------------------------------------------------------------------
// GEMM: C[ROWS,1024] = A[ROWS,1024](f16) * WT[1024,1024](f16,[n][k]) + bias
// Block tile 128x128, K-step 32, double-buffered async LDS staging.
//  OUT_MODE 0 -> f16 into [B,H,S,64]    OUT_MODE 1 -> f32 into [B,S,1024]
// ---------------------------------------------------------------------------
template <int OUT_MODE>
__global__ __launch_bounds__(256) void gemm_kernel(const _Float16* __restrict__ A,
                                                   const _Float16* __restrict__ WT,
                                                   const float* __restrict__ bias,
                                                   void* __restrict__ outp) {
  __shared__ _Float16 As[2][128][40];   // [m][k], +8 f16 pad (rows 16B aligned)
  __shared__ _Float16 Bs[2][128][40];   // [n][k]

  const int t    = threadIdx.x;
  const int wid  = t >> 5;
  const int lane = t & 31;
  const int l16  = lane & 15;
  const int hi   = lane >> 4;
  const int m0   = blockIdx.y * 128;
  const int n0   = blockIdx.x * 128;
  const int wm   = wid & 3;             // 4 wave-rows * 32 rows
  const int wn   = wid >> 2;            // 2 wave-cols * 64 cols

  CFrag acc[2][4];
#pragma unroll
  for (int im = 0; im < 2; ++im)
#pragma unroll
    for (int in = 0; in < 4; ++in)
#pragma unroll
      for (int r = 0; r < 8; ++r) acc[im][in].f[r] = 0.0f;

  auto stage = [&](int buf, int k0) {
#pragma unroll
    for (int i = 0; i < 2; ++i) {
      int idx = t + i * 256;            // 0..511, 16B chunks of the 128x32 tile
      int row = idx >> 2, c8 = idx & 3;
      cp16(&As[buf][row][c8 * 8], A  + (size_t)(m0 + row) * MODEL_DIM + k0 + c8 * 8);
      cp16(&Bs[buf][row][c8 * 8], WT + (size_t)(n0 + row) * MODEL_DIM + k0 + c8 * 8);
    }
  };

  stage(0, 0);
  const int ko = hi * 8;

  for (int kt = 0; kt < MODEL_DIM / 32; ++kt) {
    cp_wait();
    __syncthreads();                    // tile kt ready; tile kt-1 consumers done
    if (kt + 1 < MODEL_DIM / 32) stage((kt + 1) & 1, (kt + 1) * 32);
    const int buf = kt & 1;

    AFrag af[2], bf[4];
#pragma unroll
    for (int im = 0; im < 2; ++im) {
      int mr = wm * 32 + im * 16 + l16;
      af[im].h[0] = *(const f16x8*)&As[buf][mr][ko];
      af[im].h[1] = *(const f16x8*)&As[buf][mr][16 + ko];
    }
#pragma unroll
    for (int in = 0; in < 4; ++in) {
      int nr = wn * 64 + in * 16 + l16;
      bf[in].h[0] = *(const f16x8*)&Bs[buf][nr][ko];
      bf[in].h[1] = *(const f16x8*)&Bs[buf][nr][16 + ko];
    }
#pragma unroll
    for (int im = 0; im < 2; ++im)
#pragma unroll
      for (int in = 0; in < 4; ++in)
        acc[im][in].v = __builtin_amdgcn_wmma_f32_16x16x32_f16(
            false, af[im].v, false, bf[in].v, (short)0, acc[im][in].v, false, false);
  }

  float bn[4];
#pragma unroll
  for (int in = 0; in < 4; ++in) bn[in] = bias[n0 + wn * 64 + in * 16 + l16];

#pragma unroll
  for (int im = 0; im < 2; ++im)
#pragma unroll
    for (int in = 0; in < 4; ++in)
#pragma unroll
      for (int r = 0; r < 8; ++r) {
        int m = m0 + wm * 32 + im * 16 + r + 8 * hi;
        int n = n0 + wn * 64 + in * 16 + l16;
        float val = acc[im][in].f[r] + bn[in];
        if (OUT_MODE == 0) {
          int b = m >> 11, s = m & 2047;
          int h = n >> 6, d = n & 63;
          ((_Float16*)outp)[(((size_t)(b * NUM_HEAD + h) * SEQ + s) * HEAD_DIM) + d] =
              (_Float16)val;
        } else {
          ((float*)outp)[(size_t)m * MODEL_DIM + n] = val;
        }
      }
}

// ---------------------------------------------------------------------------
// Flash attention: block = 128 q rows of one (b,h); 8 waves x 16 q rows.
// K tile staged async (row-major), V tile transposed in LDS, online softmax
// across 16-lane C-row groups, P via per-wave LDS buffer (C->A layout).
// ---------------------------------------------------------------------------
__global__ __launch_bounds__(256) void attn_kernel(const _Float16* __restrict__ Q,
                                                   const _Float16* __restrict__ K,
                                                   const _Float16* __restrict__ V,
                                                   const int* __restrict__ mask,
                                                   _Float16* __restrict__ AO) {
  __shared__ _Float16 Ks[64][72];        // [key][d]
  __shared__ _Float16 VTs[64][72];       // [d][key]
  __shared__ _Float16 Ps[8][16][72];     // per-wave P buffer [m][key]

  const int t    = threadIdx.x;
  const int wid  = t >> 5;
  const int lane = t & 31;
  const int l16  = lane & 15;
  const int hi   = lane >> 4;
  const int bh   = blockIdx.y;
  const int b    = bh >> 4, h = bh & 15;
  const int q0   = blockIdx.x * 128 + wid * 16;
  const int ko   = hi * 8;

  AFrag qf[2];
  {
    const _Float16* qrow = Q + ((size_t)bh * SEQ + q0 + l16) * HEAD_DIM;
    qf[0].h[0] = *(const f16x8*)(qrow + ko);
    qf[0].h[1] = *(const f16x8*)(qrow + 16 + ko);
    qf[1].h[0] = *(const f16x8*)(qrow + 32 + ko);
    qf[1].h[1] = *(const f16x8*)(qrow + 48 + ko);
  }

  CFrag oacc[4];
  float mrow[8], lrow[8];
#pragma unroll
  for (int r = 0; r < 8; ++r) { mrow[r] = -1e30f; lrow[r] = 0.0f; }
#pragma unroll
  for (int n = 0; n < 4; ++n)
#pragma unroll
    for (int r = 0; r < 8; ++r) oacc[n].f[r] = 0.0f;

  const float scale = 0.125f;            // 1/sqrt(HEAD_DIM)

  for (int kt0 = 0; kt0 < SEQ; kt0 += 64) {
    __syncthreads();                     // previous iteration done with Ks/VTs
    // ---- stage K tile via async copies; V tile transposed on the VALU ----
#pragma unroll
    for (int i = 0; i < 2; ++i) {
      int idx = t + i * 256;             // 0..511
      int key = idx >> 3, d8 = idx & 7;
      const size_t base = ((size_t)bh * SEQ + kt0 + key) * HEAD_DIM + d8 * 8;
      cp16(&Ks[key][d8 * 8], K + base);
      f16x8 vv = *(const f16x8*)(V + base);
#pragma unroll
      for (int j = 0; j < 8; ++j) VTs[d8 * 8 + j][key] = vv[j];
    }
    cp_wait();
    __syncthreads();

    // ---- scores: 4 sub-tiles of 16 keys, 2 WMMAs each over d ----
    CFrag s[4];
#pragma unroll
    for (int j = 0; j < 4; ++j) {
      AFrag kb0, kb1;
      int kl = j * 16 + l16;
      kb0.h[0] = *(const f16x8*)&Ks[kl][ko];
      kb0.h[1] = *(const f16x8*)&Ks[kl][16 + ko];
      kb1.h[0] = *(const f16x8*)&Ks[kl][32 + ko];
      kb1.h[1] = *(const f16x8*)&Ks[kl][48 + ko];
      v8f z = {0.f, 0.f, 0.f, 0.f, 0.f, 0.f, 0.f, 0.f};
      s[j].v = __builtin_amdgcn_wmma_f32_16x16x32_f16(false, qf[0].v, false, kb0.v,
                                                      (short)0, z, false, false);
      s[j].v = __builtin_amdgcn_wmma_f32_16x16x32_f16(false, qf[1].v, false, kb1.v,
                                                      (short)0, s[j].v, false, false);
      const int keyg = kt0 + j * 16 + l16;
#pragma unroll
      for (int r = 0; r < 8; ++r) {
        int rowg = q0 + r + 8 * hi;
        int mval = mask[(size_t)rowg * SEQ + keyg];   // mask lives in 192MB L2
        s[j].f[r] = (mval == 0) ? -1e9f : s[j].f[r] * scale;
      }
    }

    // ---- online softmax ----
    float p[4][8];
#pragma unroll
    for (int r = 0; r < 8; ++r) {
      float vmax = fmaxf(fmaxf(s[0].f[r], s[1].f[r]), fmaxf(s[2].f[r], s[3].f[r]));
#pragma unroll
      for (int off = 1; off < 16; off <<= 1)
        vmax = fmaxf(vmax, __shfl_xor(vmax, off, 32));
      float mnew = fmaxf(mrow[r], vmax);
      float c = __expf(mrow[r] - mnew);
      float psum = 0.0f;
#pragma unroll
      for (int j = 0; j < 4; ++j) { p[j][r] = __expf(s[j].f[r] - mnew); psum += p[j][r]; }
#pragma unroll
      for (int off = 1; off < 16; off <<= 1)
        psum += __shfl_xor(psum, off, 32);
      lrow[r] = lrow[r] * c + psum;
      mrow[r] = mnew;
#pragma unroll
      for (int n = 0; n < 4; ++n) oacc[n].f[r] *= c;
    }

    // ---- C-layout -> A-layout for P via per-wave LDS buffer ----
#pragma unroll
    for (int j = 0; j < 4; ++j)
#pragma unroll
      for (int r = 0; r < 8; ++r)
        Ps[wid][r + 8 * hi][j * 16 + l16] = (_Float16)p[j][r];

    // ---- O += P @ V ----
#pragma unroll
    for (int kk = 0; kk < 2; ++kk) {
      AFrag pf;
      pf.h[0] = *(const f16x8*)&Ps[wid][l16][kk * 32 + ko];
      pf.h[1] = *(const f16x8*)&Ps[wid][l16][kk * 32 + 16 + ko];
#pragma unroll
      for (int n = 0; n < 4; ++n) {
        AFrag vf;
        int d = n * 16 + l16;
        vf.h[0] = *(const f16x8*)&VTs[d][kk * 32 + ko];
        vf.h[1] = *(const f16x8*)&VTs[d][kk * 32 + 16 + ko];
        oacc[n].v = __builtin_amdgcn_wmma_f32_16x16x32_f16(false, pf.v, false, vf.v,
                                                           (short)0, oacc[n].v, false, false);
      }
    }
  }

#pragma unroll
  for (int r = 0; r < 8; ++r) {
    float inv = 1.0f / lrow[r];
#pragma unroll
    for (int n = 0; n < 4; ++n) oacc[n].f[r] *= inv;
  }
#pragma unroll
  for (int n = 0; n < 4; ++n)
#pragma unroll
    for (int r = 0; r < 8; ++r) {
      int rowg = q0 + r + 8 * hi;
      size_t o = ((size_t)b * SEQ + rowg) * MODEL_DIM + h * HEAD_DIM + n * 16 + l16;
      AO[o] = (_Float16)oacc[n].f[r];
    }
}

// ---------------------------------------------------------------------------
extern "C" void kernel_launch(void* const* d_in, const int* in_sizes, int n_in,
                              void* d_out, int out_size, void* d_ws, size_t ws_size,
                              hipStream_t stream) {
  const float* q    = (const float*)d_in[0];
  const float* k    = (const float*)d_in[1];
  const float* v    = (const float*)d_in[2];
  const int*   mask = (const int*)d_in[3];
  const float* w_q  = (const float*)d_in[4];
  const float* b_q  = (const float*)d_in[5];
  const float* w_k  = (const float*)d_in[6];
  const float* b_k  = (const float*)d_in[7];
  const float* w_v  = (const float*)d_in[8];
  const float* b_v  = (const float*)d_in[9];
  const float* w_o  = (const float*)d_in[10];
  const float* b_o  = (const float*)d_in[11];

  const size_t TEN = (size_t)ROWS * MODEL_DIM;     // 4M elements
  const size_t WEL = (size_t)MODEL_DIM * MODEL_DIM;// 1M elements
  _Float16* Xh   = (_Float16*)d_ws;                // reused activation buffer
  _Float16* WTb  = Xh + TEN;                       // reused transposed weights
  _Float16* Qws  = WTb + WEL;
  _Float16* Kws  = Qws + TEN;
  _Float16* Vws  = Kws + TEN;
  _Float16* AOws = Vws + TEN;

  dim3 blk(256);
  dim3 gemm_grid(MODEL_DIM / 128, ROWS / 128);     // (8, 32)
  dim3 wt_grid(MODEL_DIM / 64, MODEL_DIM / 64);    // (16, 16)
  const int cvt_blocks = (int)(TEN / (256 * 8));   // 2048

  // Q projection
  cvt_kernel<<<cvt_blocks, blk, 0, stream>>>(q, Xh);
  wt_kernel<<<wt_grid, blk, 0, stream>>>(w_q, WTb);
  gemm_kernel<0><<<gemm_grid, blk, 0, stream>>>(Xh, WTb, b_q, Qws);
  // K projection
  cvt_kernel<<<cvt_blocks, blk, 0, stream>>>(k, Xh);
  wt_kernel<<<wt_grid, blk, 0, stream>>>(w_k, WTb);
  gemm_kernel<0><<<gemm_grid, blk, 0, stream>>>(Xh, WTb, b_k, Kws);
  // V projection
  cvt_kernel<<<cvt_blocks, blk, 0, stream>>>(v, Xh);
  wt_kernel<<<wt_grid, blk, 0, stream>>>(w_v, WTb);
  gemm_kernel<0><<<gemm_grid, blk, 0, stream>>>(Xh, WTb, b_v, Vws);
  // Attention
  attn_kernel<<<dim3(SEQ / 128, BATCH * NUM_HEAD), blk, 0, stream>>>(Qws, Kws, Vws, mask, AOws);
  // Output projection
  wt_kernel<<<wt_grid, blk, 0, stream>>>(w_o, WTb);
  gemm_kernel<1><<<gemm_grid, blk, 0, stream>>>(AOws, WTb, b_o, d_out);
}
